// AttentionBlock_57861799411800
// MI455X (gfx1250) — matrix operand order repaired and checked by reading, not compiled, not run on hardware
//
#include <hip/hip_runtime.h>
#include <math.h>

#define B_   16
#define C_   512
#define HW_  4096      // N = H*W
#define G_   32
#define CPG_ 16        // channels per group
#define NH_  8
#define HD_  64
#define O_   1536      // 3*C
#define EPS_ 1e-5f
#define LDA_ 40        // LDS row stride in halves (80B: 16B aligned, conflict-free)

typedef __attribute__((ext_vector_type(16))) _Float16 v16h;
typedef __attribute__((ext_vector_type(8)))  _Float16 v8h;
typedef __attribute__((ext_vector_type(8)))  float    v8f;

#if defined(__gfx1250__) && __has_builtin(__builtin_amdgcn_tensor_load_to_lds) && \
    __has_builtin(__builtin_amdgcn_s_wait_tensorcnt)
#define HAVE_TDM 1
#else
#define HAVE_TDM 0
#endif

__device__ __forceinline__ v16h cat8(v8h lo, v8h hi) {
  return __builtin_shufflevector(lo, hi, 0,1,2,3,4,5,6,7,8,9,10,11,12,13,14,15);
}

// A fragment (16x32 f16, MxK) from row-major [M,K] memory, row stride ld (halves).
// lane m = l&15 ; chunks at k = 8*(l>>4) and +16  (per CDNA5 ISA A layout)
__device__ __forceinline__ v16h load_a(const _Float16* t, int ld, int lane) {
  const _Float16* p = t + (size_t)(lane & 15) * ld + ((lane >> 4) << 3);
  return cat8(*(const v8h*)p, *(const v8h*)(p + 16));
}

// B fragment (32x16 f16, KxN) from memory holding B^T rows: Bm[n][k], row stride ld.
// lane n = l&15 ; chunks at k = 8*(l>>4) and +16  (same K order as A)
__device__ __forceinline__ v16h load_b(const _Float16* t, int ld, int lane) {
  const _Float16* p = t + (size_t)(lane & 15) * ld + ((lane >> 4) << 3);
  return cat8(*(const v8h*)p, *(const v8h*)(p + 16));
}

__device__ __forceinline__ v8f wmma_f16(v16h a, v16h b, v8f c) {
  return __builtin_amdgcn_wmma_f32_16x16x32_f16(false, a, false, b, (short)0, c, false, false);
}

#if HAVE_TDM
typedef unsigned int u32;
typedef u32 u32x4 __attribute__((ext_vector_type(4)));
typedef int i32x4 __attribute__((ext_vector_type(4)));
typedef int i32x8 __attribute__((ext_vector_type(8)));

// TDM: DMA a 128-row x 32-f16 (64B) tile, global row stride 512 halves (1024B),
// into LDS with 16B padding after every 64B row -> LDS row stride 80B == LDA_=40.
// Issue from ONE wave (uniform branch); complete with s_wait_tensorcnt 0.
__device__ __forceinline__ void tdm_load_tile(const void* gsrc, void* lds_dst) {
  uint64_t ga = (uint64_t)(uintptr_t)gsrc;
  u32 lds = (u32)(uintptr_t)lds_dst;      // low 32 bits of generic ptr = LDS offset
  u32x4 g0;
  g0[0] = 1u;                                          // count=1 (valid user D#)
  g0[1] = lds;                                         // lds_addr (bytes)
  g0[2] = (u32)ga;                                     // global_addr[31:0]
  g0[3] = ((u32)(ga >> 32) & 0x01FFFFFFu) | (2u << 30); // global_addr[56:32] | type=2
  i32x8 g1;
  g1[0] = (1 << 16)      // data_size = 2 bytes
        | (1 << 20)      // pad_enable
        | (3 << 22)      // pad_interval: 16 DWORDs (one 64B row)
        | (3 << 25);     // pad_amount: 4 DWORDs (16B)
  g1[1] = (32 << 16);    // tensor_dim0 = 32 elements (bits 79:48, low half)
  g1[2] = (128 << 16);   // tensor_dim1 = 128 (bits 111:80, low half)
  g1[3] = (32 << 16);    // tile_dim0 = 32 (bits 127:112)
  g1[4] = 128;           // tile_dim1 = 128 (bits 143:128); tile_dim2 = 0
  g1[5] = C_;            // tensor_dim0_stride[31:0] = 512 elements
  g1[6] = 0;             // stride0[47:32] = 0, stride1[15:0] = 0
  g1[7] = 0;
  i32x4 g2 = {0, 0, 0, 0}, g3 = {0, 0, 0, 0};          // <=2D tensor
  i32x8 g4 = {0, 0, 0, 0, 0, 0, 0, 0};                 // extra group (6-arg form)
  __builtin_amdgcn_tensor_load_to_lds(g0, g1, g2, g3, g4, 0);
}
#endif

// ---------------------------------------------------------------------------
// Kernel 1: per-(batch,group) mean / rstd.  One block reduces 16*4096 floats.
// ---------------------------------------------------------------------------
__launch_bounds__(256)
__global__ void gn_stats_kernel(const float* __restrict__ x, float* __restrict__ stats) {
  int bg = blockIdx.x;  // [0, B_*G_)
  const float4* p = (const float4*)(x + (size_t)bg * (CPG_ * HW_));
  float s = 0.f, ss = 0.f;
  for (int i = threadIdx.x; i < (CPG_ * HW_) / 4; i += 256) {
    float4 v = p[i];
    s  += v.x + v.y + v.z + v.w;
    ss += v.x * v.x + v.y * v.y + v.z * v.z + v.w * v.w;
  }
  #pragma unroll
  for (int m = 16; m > 0; m >>= 1) { s += __shfl_xor(s, m, 32); ss += __shfl_xor(ss, m, 32); }
  __shared__ float rs[8], rss[8];
  int lane = threadIdx.x & 31, wave = threadIdx.x >> 5;
  if (lane == 0) { rs[wave] = s; rss[wave] = ss; }
  __syncthreads();
  if (threadIdx.x == 0) {
    float ts = 0.f, tss = 0.f;
    for (int w = 0; w < 8; ++w) { ts += rs[w]; tss += rss[w]; }
    const float inv = 1.0f / (float)(CPG_ * HW_);
    float mean = ts * inv;
    float var  = tss * inv - mean * mean;
    stats[bg * 2 + 0] = mean;
    stats[bg * 2 + 1] = rsqrtf(var + EPS_);
  }
}

// ---------------------------------------------------------------------------
// Kernel 2: convert both weight matrices fp32 -> f16.
// ---------------------------------------------------------------------------
__launch_bounds__(256)
__global__ void cvt_weights_kernel(const float* __restrict__ wq, const float* __restrict__ wp,
                                   _Float16* __restrict__ wqh, _Float16* __restrict__ wph) {
  int i = blockIdx.x * 256 + threadIdx.x;
  if (i < O_ * C_) wqh[i] = (_Float16)wq[i];
  if (i < C_ * C_) wph[i] = (_Float16)wp[i];
}

// ---------------------------------------------------------------------------
// Kernel 3: fused GroupNorm-apply + QKV GEMM.
//   y[b,o,n] = sum_c Wq[o,c] * xn[b,c,n] + bq[o]
//   A tile (weights) staged by TDM; B tile staged manually with fused GroupNorm.
//   q,k stored [b][o][n] row-major; v stored transposed per head: [b][h][n][d]
//   Block tile 128x128, 8 waves (4x2), wave tile 32x64, K-step 32.
// ---------------------------------------------------------------------------
__launch_bounds__(256)
__global__ void qkv_gemm_kernel(const float* __restrict__ x, const float* __restrict__ stats,
                                const float* __restrict__ gamma, const float* __restrict__ beta,
                                const _Float16* __restrict__ wq, const float* __restrict__ bq,
                                _Float16* __restrict__ qkv) {
  __shared__ __attribute__((aligned(16))) _Float16 As[128 * LDA_];
  __shared__ __attribute__((aligned(16))) _Float16 Bs[128 * LDA_];
  int bidx = blockIdx.x;
  int nb = bidx & 31;
  int t2 = bidx >> 5;
  int mb = t2 % 12;
  int b  = t2 / 12;
  int mo = mb * 128, no = nb * 128;
  int t = threadIdx.x, lane = t & 31, wave = t >> 5;
  int wm = wave & 3, wn = wave >> 2;   // 4 (M) x 2 (N) waves
  v8f acc[2][4] = {};
  int nl = t & 127, cg = t >> 7;       // B-loader: pixel, 16-channel half

  for (int k = 0; k < C_; k += 32) {
    // ---- stage A tile: wq rows [mo,mo+128) x cols [k,k+32)
#if HAVE_TDM
    if (wave == 0)
      tdm_load_tile(&wq[(size_t)mo * C_ + k], As);
#else
    #pragma unroll
    for (int i = 0; i < 2; ++i) {
      int cidx = t + 256 * i;
      int r = cidx >> 2, kc = (cidx & 3) << 3;
      *(uint4*)&As[r * LDA_ + kc] = *(const uint4*)&wq[(size_t)(mo + r) * C_ + k + kc];
    }
#endif
    // ---- stage B tile with fused GroupNorm: pixels [no,no+128) x channels [k,k+32)
    {
      int cbase = k + cg * 16;           // 16-channel slab lies in one group
      int g = cbase >> 4;
      float mean = stats[(b * G_ + g) * 2 + 0];
      float rstd = stats[(b * G_ + g) * 2 + 1];
      const float* xp = x + ((size_t)b * C_ + cbase) * HW_ + no + nl;
      if (k + 32 < C_)
        __builtin_prefetch(xp + (size_t)32 * HW_, 0, 0);   // next K slab of x
      v8h t0, t1;
      #pragma unroll
      for (int j = 0; j < 8; ++j) {
        float v0 = xp[(size_t)j * HW_];
        float v1 = xp[(size_t)(j + 8) * HW_];
        t0[j] = (_Float16)((v0 - mean) * rstd * gamma[cbase + j] + beta[cbase + j]);
        t1[j] = (_Float16)((v1 - mean) * rstd * gamma[cbase + j + 8] + beta[cbase + j + 8]);
      }
      *(v8h*)&Bs[nl * LDA_ + cg * 16]     = t0;
      *(v8h*)&Bs[nl * LDA_ + cg * 16 + 8] = t1;
    }
#if HAVE_TDM
    if (wave == 0) __builtin_amdgcn_s_wait_tensorcnt((short)0);
#endif
    __syncthreads();

    v16h af[2], bf[4];
    #pragma unroll
    for (int i = 0; i < 2; ++i) af[i] = load_a(&As[(wm * 32 + i * 16) * LDA_], LDA_, lane);
    #pragma unroll
    for (int j = 0; j < 4; ++j) bf[j] = load_b(&Bs[(wn * 64 + j * 16) * LDA_], LDA_, lane);
    #pragma unroll
    for (int i = 0; i < 2; ++i)
      #pragma unroll
      for (int j = 0; j < 4; ++j)
        acc[i][j] = wmma_f16(af[i], bf[j], acc[i][j]);
    __syncthreads();
  }

  // ---- epilogue: + bias, f16 store. q/k row-major [o][n]; v transposed [h][n][d].
  int mbase = mo + wm * 32 + 8 * (lane >> 4);
  int nbase = no + wn * 64 + (lane & 15);
  #pragma unroll
  for (int i = 0; i < 2; ++i) {
    #pragma unroll
    for (int j = 0; j < 4; ++j) {
      int nt = nbase + j * 16;
      #pragma unroll
      for (int r = 0; r < 8; ++r) {
        int o = mbase + i * 16 + r;
        float val = acc[i][j][r] + bq[o];
        if (o < 2 * C_) {
          qkv[((size_t)b * O_ + o) * HW_ + nt] = (_Float16)val;
        } else {
          int oc = o - 2 * C_;
          int hh = oc >> 6, dd = oc & 63;
          qkv[((size_t)b * O_ + 2 * C_) * HW_ + (size_t)hh * (HW_ * HD_) +
              (size_t)nt * HD_ + dd] = (_Float16)val;
        }
      }
    }
  }
}

// ---------------------------------------------------------------------------
// Kernel 4: S = scale * q k^T  (64x64 per (b,h), K = 4096), softmax rows -> P f16.
// 8 waves: wave w computes row-block (w&3) over K-half (w>>2); upper-half
// partials are combined through LDS, lower-half waves finish softmax.
// ---------------------------------------------------------------------------
__launch_bounds__(256)
__global__ void attn_softmax_kernel(const _Float16* __restrict__ qkv, _Float16* __restrict__ P) {
  int bh = blockIdx.x;                      // [0,128)
  int b = bh >> 3, hh = bh & 7;
  int lane = threadIdx.x & 31, wave = threadIdx.x >> 5;
  int rb = wave & 3;                        // row block (16 rows)
  int kh = wave >> 2;                       // K half
  const _Float16* q  = qkv + ((size_t)b * O_ + hh * HD_) * HW_ + (size_t)(rb * 16) * HW_;
  const _Float16* kp = qkv + ((size_t)b * O_ + C_ + hh * HD_) * HW_;
  v8f acc[4] = {};
  int n0 = kh * (HW_ / 2);
  for (int n = n0; n < n0 + HW_ / 2; n += 32) {
    v16h af = load_a(q + n, HW_, lane);
    #pragma unroll
    for (int j = 0; j < 4; ++j) {
      v16h bf = load_b(kp + (size_t)(j * 16) * HW_ + n, HW_, lane);
      acc[j] = wmma_f16(af, bf, acc[j]);
    }
  }
  __shared__ float Sred[4][4][8][32];       // [rowblock][tile j][vgpr r][lane] : 16 KB
  if (kh == 1) {
    #pragma unroll
    for (int j = 0; j < 4; ++j)
      #pragma unroll
      for (int r = 0; r < 8; ++r)
        Sred[rb][j][r][lane] = acc[j][r];
  }
  __syncthreads();
  if (kh == 0) {
    const float scale = 0.125f;  // 64^-0.5
    int row = rb * 16 + 8 * (lane >> 4);
    int col = lane & 15;
    #pragma unroll
    for (int r = 0; r < 8; ++r) {
      float v0 = (acc[0][r] + Sred[rb][0][r][lane]) * scale;
      float v1 = (acc[1][r] + Sred[rb][1][r][lane]) * scale;
      float v2 = (acc[2][r] + Sred[rb][2][r][lane]) * scale;
      float v3 = (acc[3][r] + Sred[rb][3][r][lane]) * scale;
      float mx = fmaxf(fmaxf(v0, v1), fmaxf(v2, v3));
      #pragma unroll
      for (int m = 8; m > 0; m >>= 1) mx = fmaxf(mx, __shfl_xor(mx, m, 32)); // 16-lane halves
      float e0 = __expf(v0 - mx), e1 = __expf(v1 - mx);
      float e2 = __expf(v2 - mx), e3 = __expf(v3 - mx);
      float sm = e0 + e1 + e2 + e3;
      #pragma unroll
      for (int m = 8; m > 0; m >>= 1) sm += __shfl_xor(sm, m, 32);
      float inv = 1.0f / sm;
      size_t base = ((size_t)bh * HD_ + row + r) * HD_ + col;
      P[base + 0]  = (_Float16)(e0 * inv);
      P[base + 16] = (_Float16)(e1 * inv);
      P[base + 32] = (_Float16)(e2 * inv);
      P[base + 48] = (_Float16)(e3 * inv);
    }
  }
}

// ---------------------------------------------------------------------------
// Kernel 5: out = P @ v, stored pixel-major [b][n][c] (c = h*64+d), aliased over
// the q region of the qkv buffer (q already consumed). Block: (b,h) x 256 pixels.
// ---------------------------------------------------------------------------
__launch_bounds__(256)
__global__ void pv_gemm_kernel(const _Float16* __restrict__ P, _Float16* __restrict__ qkv) {
  int bid = blockIdx.x;                // 128 * 16
  int nb = bid & 15;
  int bh = bid >> 4;
  int b = bh >> 3, hh = bh & 7;
  int no = nb * 256;
  int t = threadIdx.x, lane = t & 31, wave = t >> 5;
  int wm = wave & 1, wn = wave >> 1;   // 2 (M=64) x 4 (N=256) waves, wave tile 32x64
  const _Float16* Pm = P + (size_t)bh * (HD_ * HD_);
  const _Float16* vt = qkv + ((size_t)b * O_ + 2 * C_) * HW_ + (size_t)hh * (HW_ * HD_);
  v8f acc[2][4] = {};
  #pragma unroll
  for (int k = 0; k < HD_; k += 32) {
    v16h af[2], bf[4];
    #pragma unroll
    for (int i = 0; i < 2; ++i)
      af[i] = load_a(Pm + (size_t)(wm * 32 + i * 16) * HD_ + k, HD_, lane);
    #pragma unroll
    for (int j = 0; j < 4; ++j)
      bf[j] = load_b(vt + (size_t)(no + wn * 64 + j * 16) * HD_ + k, HD_, lane);
    #pragma unroll
    for (int i = 0; i < 2; ++i)
      #pragma unroll
      for (int j = 0; j < 4; ++j)
        acc[i][j] = wmma_f16(af[i], bf[j], acc[i][j]);
  }
  _Float16* attnout = qkv + (size_t)b * O_ * HW_;   // alias: q region of batch b
  int ddbase = wm * 32 + 8 * (lane >> 4);
  int nbase  = no + wn * 64 + (lane & 15);
  #pragma unroll
  for (int i = 0; i < 2; ++i)
    #pragma unroll
    for (int j = 0; j < 4; ++j)
      #pragma unroll
      for (int r = 0; r < 8; ++r) {
        int dd = ddbase + i * 16 + r;
        int n  = nbase + j * 16;
        attnout[(size_t)n * C_ + hh * HD_ + dd] = (_Float16)acc[i][j][r];
      }
}

// ---------------------------------------------------------------------------
// Kernel 6: proj GEMM + bias + residual, fp32 out [b][c][n].
// Both A and B tiles are plain strided 2D copies -> staged by TDM.
// ---------------------------------------------------------------------------
__launch_bounds__(256)
__global__ void proj_gemm_kernel(const _Float16* __restrict__ wp, const _Float16* __restrict__ qkv,
                                 const float* __restrict__ bp, const float* __restrict__ x,
                                 float* __restrict__ out) {
  __shared__ __attribute__((aligned(16))) _Float16 As[128 * LDA_];
  __shared__ __attribute__((aligned(16))) _Float16 Bs[128 * LDA_];
  int bidx = blockIdx.x;               // 16 * 4 * 32
  int nb = bidx & 31, mb = (bidx >> 5) & 3, b = bidx >> 7;
  int mo = mb * 128, no = nb * 128;
  int t = threadIdx.x, lane = t & 31, wave = t >> 5;
  int wm = wave & 3, wn = wave >> 2;
  const _Float16* attnout = qkv + (size_t)b * O_ * HW_;
  v8f acc[2][4] = {};
  for (int k = 0; k < C_; k += 32) {
#if HAVE_TDM
    if (wave == 0) {
      tdm_load_tile(&wp[(size_t)mo * C_ + k], As);
      tdm_load_tile(&attnout[(size_t)no * C_ + k], Bs);
      __builtin_amdgcn_s_wait_tensorcnt((short)0);
    }
#else
    #pragma unroll
    for (int i = 0; i < 2; ++i) {
      int cidx = t + 256 * i;
      int r = cidx >> 2, kc = (cidx & 3) << 3;
      *(uint4*)&As[r * LDA_ + kc] = *(const uint4*)&wp[(size_t)(mo + r) * C_ + k + kc];
      *(uint4*)&Bs[r * LDA_ + kc] = *(const uint4*)&attnout[(size_t)(no + r) * C_ + k + kc];
    }
#endif
    __syncthreads();
    v16h af[2], bf[4];
    #pragma unroll
    for (int i = 0; i < 2; ++i) af[i] = load_a(&As[(wm * 32 + i * 16) * LDA_], LDA_, lane);
    #pragma unroll
    for (int j = 0; j < 4; ++j) bf[j] = load_b(&Bs[(wn * 64 + j * 16) * LDA_], LDA_, lane);
    #pragma unroll
    for (int i = 0; i < 2; ++i)
      #pragma unroll
      for (int j = 0; j < 4; ++j)
        acc[i][j] = wmma_f16(af[i], bf[j], acc[i][j]);
    __syncthreads();
  }
  int mbase = mo + wm * 32 + 8 * (lane >> 4);
  int nbase = no + wn * 64 + (lane & 15);
  #pragma unroll
  for (int i = 0; i < 2; ++i)
    #pragma unroll
    for (int j = 0; j < 4; ++j) {
      int nt = nbase + j * 16;
      #pragma unroll
      for (int r = 0; r < 8; ++r) {
        int o = mbase + i * 16 + r;
        size_t idx = ((size_t)b * C_ + o) * HW_ + nt;
        out[idx] = acc[i][j][r] + bp[o] + x[idx];
      }
    }
}

// ---------------------------------------------------------------------------
extern "C" void kernel_launch(void* const* d_in, const int* in_sizes, int n_in,
                              void* d_out, int out_size, void* d_ws, size_t ws_size,
                              hipStream_t stream) {
  const float* x      = (const float*)d_in[0];
  const float* gamma  = (const float*)d_in[1];
  const float* beta   = (const float*)d_in[2];
  const float* w_qkv  = (const float*)d_in[3];
  const float* b_qkv  = (const float*)d_in[4];
  const float* w_proj = (const float*)d_in[5];
  const float* b_proj = (const float*)d_in[6];
  float* out = (float*)d_out;

  char* ws = (char*)d_ws;
  float* stats  = (float*)ws;      ws += (size_t)B_ * G_ * 2 * sizeof(float);   // 4 KB
  _Float16* wqh = (_Float16*)ws;   ws += (size_t)O_ * C_ * sizeof(_Float16);    // 1.5 MB
  _Float16* wph = (_Float16*)ws;   ws += (size_t)C_ * C_ * sizeof(_Float16);    // 0.5 MB
  _Float16* P   = (_Float16*)ws;   ws += (size_t)B_ * NH_ * HD_ * HD_ * sizeof(_Float16); // 1 MB
  _Float16* qkv = (_Float16*)ws;   // B_*O_*HW_ halves = 192 MB (q region reused for attn-out)

  gn_stats_kernel<<<B_ * G_, 256, 0, stream>>>(x, stats);
  cvt_weights_kernel<<<(O_ * C_ + 255) / 256, 256, 0, stream>>>(w_qkv, w_proj, wqh, wph);
  qkv_gemm_kernel<<<B_ * 12 * 32, 256, 0, stream>>>(x, stats, gamma, beta, wqh, b_qkv, qkv);
  attn_softmax_kernel<<<B_ * NH_, 256, 0, stream>>>(qkv, P);
  pv_gemm_kernel<<<B_ * NH_ * 16, 256, 0, stream>>>(P, qkv);
  proj_gemm_kernel<<<B_ * 4 * 32, 256, 0, stream>>>(wph, qkv, b_proj, x, out);
}
